// TrajectoryPredictor_23132693856841
// MI455X (gfx1250) — compile-verified, matrix operand order checked by reading, not run on hardware
//
#include <hip/hip_runtime.h>
#include <hip/hip_bf16.h>

typedef __attribute__((ext_vector_type(16))) _Float16 v16h;
typedef __attribute__((ext_vector_type(8)))  _Float16 v8h;
typedef __attribute__((ext_vector_type(8)))  float    v8f;
typedef __attribute__((ext_vector_type(2)))  float    f32x2;

#define NB   65536
#define TT   15
#define HH   128
#define NIN  6
#define NOUT 3
#define HOR  10
#define G4   512   // 4*H

// ---- LDS layout (bytes) ----
#define OFF_G     0        // float[16*512]    gates
#define OFF_WIH   32768    // f16[32][4][32][16]  W_ih B-fragments
#define OFF_H0    163840   // f16[15][16][128] layer0 outputs
#define OFF_HA    225280   // f16[16][128]     current h
#define OFF_DA    229376   // f16[16][128]     decoder input
#define OFF_BIAS  233472   // float[512]
#define OFF_LNG   235520   // float[128]
#define OFF_LNB   236032   // float[128]
#define OFF_BASE  236544   // float[10*128]
#define OFF_PREV  241664   // float[16*3] (padded)
#define OFF_WM    241920   // float[384]
#define OFF_WS    243456   // float[384]
#define OFF_WP3   244992   // float[384]
#define OFF_XA    246528   // f16[15][16][8]
#define LDS_BYTES 250368

__device__ __forceinline__ float sigm(float v) { return 1.f / (1.f + __expf(-v)); }
__device__ __forceinline__ float tanh_f(float v) {
  float a = fabsf(v);
  float e = __expf(-2.f * a);
  float t = (1.f - e) / (1.f + e);
  return v < 0.f ? -t : t;
}

__device__ __forceinline__ v8f wmma_f16(v16h a, v16h b, v8f c) {
  // v_wmma_f32_16x16x32_f16: (neg_a, A, neg_b, B, c_mod, C, reuse_a, reuse_b)
  return __builtin_amdgcn_wmma_f32_16x16x32_f16(false, a, false, b, (short)0, c, false, false);
}

// Stage W_ih (row-major [512][K]) into LDS in B-fragment layout:
// wih[tile][chunk][lane][16 f16], k = chunk*32 + ((e>>3)<<4) + ((lane>>4)<<3) + (e&7)
__device__ __forceinline__ void stage_wih(_Float16* wih, const float* __restrict__ W,
                                          int K, int nchunks, int tid) {
  int total = 32 * nchunks * 256;
  for (int p = tid; p < total; p += 256) {
    int tc   = p >> 8;
    int tile = tc / nchunks;
    int chunk = tc - tile * nchunks;
    int lane = (p >> 3) & 31;
    int e    = (p & 7) * 2;
    int n    = lane & 15;
    int k    = chunk * 32 + ((e >> 3) << 4) + ((lane >> 4) << 3) + (e & 7);
    int row  = tile * 16 + n;
    float w0 = (k     < K) ? W[row * K + k]     : 0.f;
    float w1 = (k + 1 < K) ? W[row * K + k + 1] : 0.f;
    int idx  = (((tile << 2) + chunk) * 32 + lane) * 16 + e;
    wih[idx]     = (_Float16)w0;
    wih[idx + 1] = (_Float16)w1;
  }
}

// Per-wave W_hh slice (rows wave*64 .. +63 of [512][128]) as 16 register B-fragments.
__device__ __forceinline__ void load_whh(v16h* whh, const float* __restrict__ W,
                                         int wave, int lane) {
  int n = lane & 15;
  int koff = (lane >> 4) << 3;
#pragma unroll
  for (int i = 0; i < 4; ++i) {
    const float* rp = W + ((wave * 4 + i) * 16 + n) * 128;
#pragma unroll
    for (int kc = 0; kc < 4; ++kc) {
      v16h f;
#pragma unroll
      for (int ep = 0; ep < 8; ++ep) {
        int e = ep * 2;
        int k = kc * 32 + ((e >> 3) << 4) + koff + (e & 7);
        f32x2 w = *(const f32x2*)(rp + k);
        f[e]     = (_Float16)w.x;
        f[e + 1] = (_Float16)w.y;
      }
      whh[i * 4 + kc] = f;
    }
  }
}

// A-fragment (16x32 f16) from LDS row-major f16[16][128]
__device__ __forceinline__ v16h load_a(const _Float16* src, int lane, int kc) {
  int m = lane & 15;
  const _Float16* p = src + m * 128 + kc * 32 + ((lane >> 4) << 3);
  v8h lo = *(const v8h*)p;
  v8h hi = *(const v8h*)(p + 16);
  return __builtin_shufflevector(lo, hi, 0,1,2,3,4,5,6,7,8,9,10,11,12,13,14,15);
}

// B-fragment from staged LDS layout
__device__ __forceinline__ v16h load_b(const _Float16* wih, int tile, int kc, int lane) {
  const _Float16* p = wih + (((tile << 2) + kc) * 32 + lane) * 16;
  v8h lo = *(const v8h*)p;
  v8h hi = *(const v8h*)(p + 8);
  return __builtin_shufflevector(lo, hi, 0,1,2,3,4,5,6,7,8,9,10,11,12,13,14,15);
}

// Scatter a 16x16 f32 D tile into gates LDS (float[16][512])
__device__ __forceinline__ void store_gates(float* g, int lane, int tileN, v8f acc) {
  int n = lane & 15;
  int half = lane >> 4;
  int col = tileN * 16 + n;
#pragma unroll
  for (int q = 0; q < 8; ++q) g[(q + half * 8) * 512 + col] = acc[q];
}

__global__ __launch_bounds__(256) void traj_kernel(
    const float* __restrict__ x,
    const float* __restrict__ Wih0, const float* __restrict__ Whh0, const float* __restrict__ b0,
    const float* __restrict__ Wih1, const float* __restrict__ Whh1, const float* __restrict__ b1,
    const float* __restrict__ Wihd, const float* __restrict__ Whhd, const float* __restrict__ bd,
    const float* __restrict__ emb, const float* __restrict__ Wp, const float* __restrict__ bp,
    const float* __restrict__ Wm,  const float* __restrict__ bm,
    const float* __restrict__ Wsg, const float* __restrict__ bs,
    const float* __restrict__ lng_g, const float* __restrict__ lnb_g,
    float* __restrict__ out) {
  extern __shared__ char smem[];
  float*    gls   = (float*)(smem + OFF_G);
  _Float16* wih   = (_Float16*)(smem + OFF_WIH);
  _Float16* h0s   = (_Float16*)(smem + OFF_H0);
  _Float16* hA    = (_Float16*)(smem + OFF_HA);
  _Float16* dA    = (_Float16*)(smem + OFF_DA);
  float*    biasl = (float*)(smem + OFF_BIAS);
  float*    lng   = (float*)(smem + OFF_LNG);
  float*    lnb   = (float*)(smem + OFF_LNB);
  float*    baseb = (float*)(smem + OFF_BASE);
  float*    prevb = (float*)(smem + OFF_PREV);
  float*    wml   = (float*)(smem + OFF_WM);
  float*    wsl   = (float*)(smem + OFF_WS);
  float*    wp3   = (float*)(smem + OFF_WP3);
  _Float16* xa    = (_Float16*)(smem + OFF_XA);

  const int tid  = threadIdx.x;
  const int lane = tid & 31;
  const int wave = tid >> 5;
  const int r    = tid >> 4;         // elementwise row 0..15
  const int c0   = (tid & 15) * 8;   // elementwise col group
  const int base16 = blockIdx.x * 16;

  // ================= phase A staging =================
  for (int p = tid; p < TT * 16 * NIN; p += 256) {
    int t = p / (16 * NIN);
    int rem = p - t * (16 * NIN);
    int m = rem / NIN;
    int i = rem - m * NIN;
    xa[t * 128 + m * 8 + i] = (_Float16)x[(size_t)(base16 + m) * TT * NIN + t * NIN + i];
  }
  for (int p = tid; p < TT * 16 * 2; p += 256) {
    int t = p >> 5; int rem = p & 31; int m = rem >> 1; int i = rem & 1;
    xa[t * 128 + m * 8 + 6 + i] = (_Float16)0.f;
  }
  for (int p = tid; p < 16 * 128; p += 256) hA[p] = (_Float16)0.f;
  for (int p = tid; p < G4; p += 256) biasl[p] = b0[p];
  stage_wih(wih, Wih0, NIN, 1, tid);   // K=6 padded into chunk 0

  v16h whh[16];
  load_whh(whh, Whh0, wave, lane);

  float cst[8];
#pragma unroll
  for (int j = 0; j < 8; ++j) cst[j] = 0.f;

  // ================= phase A: LSTM layer 0 =================
#pragma unroll 1
  for (int t = 0; t < TT; ++t) {
    __syncthreads();
    v8f acc[4];
#pragma unroll
    for (int i = 0; i < 4; ++i)
#pragma unroll
      for (int q = 0; q < 8; ++q) acc[i][q] = 0.f;

    v8h zz;
#pragma unroll
    for (int q = 0; q < 8; ++q) zz[q] = (_Float16)0.f;
    v8h xlo = zz;
    if (lane < 16) xlo = *(const v8h*)(xa + t * 128 + lane * 8);
    v16h ax = __builtin_shufflevector(xlo, zz, 0,1,2,3,4,5,6,7,8,9,10,11,12,13,14,15);
#pragma unroll
    for (int i = 0; i < 4; ++i)
      acc[i] = wmma_f16(ax, load_b(wih, wave * 4 + i, 0, lane), acc[i]);

#pragma unroll
    for (int kc = 0; kc < 4; ++kc) {
      v16h ah = load_a(hA, lane, kc);
#pragma unroll
      for (int i = 0; i < 4; ++i)
        acc[i] = wmma_f16(ah, whh[i * 4 + kc], acc[i]);
    }
#pragma unroll
    for (int i = 0; i < 4; ++i) store_gates(gls, lane, wave * 4 + i, acc[i]);
    __syncthreads();

    const float* gr = gls + r * 512;
#pragma unroll
    for (int j = 0; j < 8; ++j) {
      int col = c0 + j;
      float gi = gr[col]        + biasl[col];
      float gf = gr[128 + col]  + biasl[128 + col];
      float gg = gr[256 + col]  + biasl[256 + col];
      float go = gr[384 + col]  + biasl[384 + col];
      cst[j] = sigm(gf) * cst[j] + sigm(gi) * tanh_f(gg);
      float hvv = sigm(go) * tanh_f(cst[j]);
      _Float16 h16 = (_Float16)hvv;
      hA[r * 128 + col] = h16;
      h0s[t * 2048 + r * 128 + col] = h16;
    }
  }

  // ================= phase B: LSTM layer 1 =================
  __syncthreads();
  stage_wih(wih, Wih1, HH, 4, tid);
  load_whh(whh, Whh1, wave, lane);
  for (int p = tid; p < G4; p += 256) biasl[p] = b1[p];
  for (int p = tid; p < 16 * 128; p += 256) hA[p] = (_Float16)0.f;
#pragma unroll
  for (int j = 0; j < 8; ++j) cst[j] = 0.f;

#pragma unroll 1
  for (int t = 0; t < TT; ++t) {
    __syncthreads();
    v8f acc[4];
#pragma unroll
    for (int i = 0; i < 4; ++i)
#pragma unroll
      for (int q = 0; q < 8; ++q) acc[i][q] = 0.f;

#pragma unroll
    for (int kc = 0; kc < 4; ++kc) {
      v16h ain = load_a(h0s + t * 2048, lane, kc);
      v16h ah  = load_a(hA, lane, kc);
#pragma unroll
      for (int i = 0; i < 4; ++i) {
        acc[i] = wmma_f16(ain, load_b(wih, wave * 4 + i, kc, lane), acc[i]);
        acc[i] = wmma_f16(ah, whh[i * 4 + kc], acc[i]);
      }
    }
#pragma unroll
    for (int i = 0; i < 4; ++i) store_gates(gls, lane, wave * 4 + i, acc[i]);
    __syncthreads();

    const float* gr = gls + r * 512;
#pragma unroll
    for (int j = 0; j < 8; ++j) {
      int col = c0 + j;
      float gi = gr[col]        + biasl[col];
      float gf = gr[128 + col]  + biasl[128 + col];
      float gg = gr[256 + col]  + biasl[256 + col];
      float go = gr[384 + col]  + biasl[384 + col];
      cst[j] = sigm(gf) * cst[j] + sigm(gi) * tanh_f(gg);
      float hvv = sigm(go) * tanh_f(cst[j]);
      hA[r * 128 + col] = (_Float16)hvv;
    }
  }

  // ================= phase C: decoder =================
  __syncthreads();
  stage_wih(wih, Wihd, HH, 4, tid);
  load_whh(whh, Whhd, wave, lane);
  for (int p = tid; p < G4; p += 256) biasl[p] = bd[p];
  if (tid < 128) { lng[tid] = lng_g[tid]; lnb[tid] = lnb_g[tid]; }
  for (int p = tid; p < 384; p += 256) { wml[p] = Wm[p]; wsl[p] = Wsg[p]; }
  for (int p = tid; p < 384; p += 256) {
    int col = p / 3, o = p - col * 3;
    wp3[p] = Wp[col * 131 + o];
  }
  for (int p = tid; p < HOR * 128; p += 256) {
    int s = p >> 7; int j = p & 127;
    float a = bp[j];
    const float* wr = Wp + j * 131 + 3;
    const float* er = emb + s * 128;
#pragma unroll 4
    for (int k = 0; k < 128; ++k) a += er[k] * wr[k];
    baseb[p] = a;
  }
  if (tid < 48) {
    int rr = tid / 3, o = tid - rr * 3;
    prevb[tid] = x[(size_t)(base16 + rr) * TT * NIN + 14 * NIN + o];
  }

  const size_t lsoff = (size_t)NB * HOR * NOUT;
#pragma unroll 1
  for (int s = 0; s < HOR; ++s) {
    __syncthreads();
    {
      float p0 = prevb[r * 3 + 0], p1 = prevb[r * 3 + 1], p2 = prevb[r * 3 + 2];
#pragma unroll
      for (int j = 0; j < 8; ++j) {
        int col = c0 + j;
        float v = baseb[s * 128 + col] + p0 * wp3[col * 3 + 0] +
                  p1 * wp3[col * 3 + 1] + p2 * wp3[col * 3 + 2];
        dA[r * 128 + col] = (_Float16)fmaxf(v, 0.f);
      }
    }
    __syncthreads();

    v8f acc[4];
#pragma unroll
    for (int i = 0; i < 4; ++i)
#pragma unroll
      for (int q = 0; q < 8; ++q) acc[i][q] = 0.f;
#pragma unroll
    for (int kc = 0; kc < 4; ++kc) {
      v16h ain = load_a(dA, lane, kc);
      v16h ah  = load_a(hA, lane, kc);
#pragma unroll
      for (int i = 0; i < 4; ++i) {
        acc[i] = wmma_f16(ain, load_b(wih, wave * 4 + i, kc, lane), acc[i]);
        acc[i] = wmma_f16(ah, whh[i * 4 + kc], acc[i]);
      }
    }
#pragma unroll
    for (int i = 0; i < 4; ++i) store_gates(gls, lane, wave * 4 + i, acc[i]);
    __syncthreads();

    float hv[8];
    const float* gr = gls + r * 512;
#pragma unroll
    for (int j = 0; j < 8; ++j) {
      int col = c0 + j;
      float gi = gr[col]        + biasl[col];
      float gf = gr[128 + col]  + biasl[128 + col];
      float gg = gr[256 + col]  + biasl[256 + col];
      float go = gr[384 + col]  + biasl[384 + col];
      cst[j] = sigm(gf) * cst[j] + sigm(gi) * tanh_f(gg);
      hv[j] = sigm(go) * tanh_f(cst[j]);
      hA[r * 128 + col] = (_Float16)hv[j];
    }

    // layernorm over the 16 threads of this row (half-wave lane group)
    float sum = 0.f, sq = 0.f;
#pragma unroll
    for (int j = 0; j < 8; ++j) { sum += hv[j]; sq += hv[j] * hv[j]; }
#pragma unroll
    for (int m = 1; m < 16; m <<= 1) {
      sum += __shfl_xor(sum, m, 16);
      sq  += __shfl_xor(sq, m, 16);
    }
    float mean = sum * (1.f / 128.f);
    float var  = sq * (1.f / 128.f) - mean * mean;
    float rstd = rsqrtf(var + 1e-5f);

    float pm[3] = {0.f, 0.f, 0.f}, ps[3] = {0.f, 0.f, 0.f};
#pragma unroll
    for (int j = 0; j < 8; ++j) {
      int col = c0 + j;
      float fv = (hv[j] - mean) * rstd * lng[col] + lnb[col];
#pragma unroll
      for (int o = 0; o < 3; ++o) {
        pm[o] += fv * wml[o * 128 + col];
        ps[o] += fv * wsl[o * 128 + col];
      }
    }
#pragma unroll
    for (int m = 1; m < 16; m <<= 1)
#pragma unroll
      for (int o = 0; o < 3; ++o) {
        pm[o] += __shfl_xor(pm[o], m, 16);
        ps[o] += __shfl_xor(ps[o], m, 16);
      }

    if ((tid & 15) == 0) {
      size_t ob = ((size_t)(base16 + r) * HOR + s) * NOUT;
#pragma unroll
      for (int o = 0; o < 3; ++o) {
        float mu = pm[o] + bm[o];
        float ls = fminf(3.f, fmaxf(-6.f, ps[o] + bs[o]));
        out[ob + o] = mu;
        out[lsoff + ob + o] = ls;
        prevb[r * 3 + o] = mu;
      }
    }
  }
}

extern "C" void kernel_launch(void* const* d_in, const int* in_sizes, int n_in,
                              void* d_out, int out_size, void* d_ws, size_t ws_size,
                              hipStream_t stream) {
  (void)in_sizes; (void)n_in; (void)d_ws; (void)ws_size; (void)out_size;
  const float* x    = (const float*)d_in[0];
  const float* Wih0 = (const float*)d_in[1];
  const float* Whh0 = (const float*)d_in[2];
  const float* b0   = (const float*)d_in[3];
  const float* Wih1 = (const float*)d_in[4];
  const float* Whh1 = (const float*)d_in[5];
  const float* b1   = (const float*)d_in[6];
  const float* Wihd = (const float*)d_in[7];
  const float* Whhd = (const float*)d_in[8];
  const float* bd   = (const float*)d_in[9];
  const float* emb  = (const float*)d_in[10];
  const float* Wp   = (const float*)d_in[11];
  const float* bp   = (const float*)d_in[12];
  const float* Wm   = (const float*)d_in[13];
  const float* bm   = (const float*)d_in[14];
  const float* Wsg  = (const float*)d_in[15];
  const float* bs   = (const float*)d_in[16];
  const float* lng  = (const float*)d_in[17];
  const float* lnb  = (const float*)d_in[18];
  float* out = (float*)d_out;

  traj_kernel<<<dim3(NB / 16), dim3(256), LDS_BYTES, stream>>>(
      x, Wih0, Whh0, b0, Wih1, Whh1, b1, Wihd, Whhd, bd,
      emb, Wp, bp, Wm, bm, Wsg, bs, lng, lnb, out);
}
